// MomentumQueueClass_17162689315190
// MI455X (gfx1250) — matrix operand order repaired and checked by reading, not compiled, not run on hardware
//
#include <hip/hip_runtime.h>

typedef __attribute__((ext_vector_type(16))) _Float16 v16h;
typedef __attribute__((ext_vector_type(8)))  _Float16 v8h;
typedef __attribute__((ext_vector_type(4)))  _Float16 v4h;
typedef __attribute__((ext_vector_type(8)))  float    v8f;

#define B_N   1024
#define D_K   512
#define Q_N   65536
#define K_TOP 32
#define C_N   345
#define INV_T (1.0f / 0.07f)

#define N_CHUNKS        32
#define CHUNK_COLS      (Q_N / N_CHUNKS)        /* 2048 */
#define TILE_N          256
#define TILES_PER_CHUNK (CHUNK_COLS / TILE_N)   /* 8 */
#define BM              64
#define CAND_PER_ROW    (N_CHUNKS * K_TOP)      /* 1024 */

#define NEG_INF (-__builtin_inff())

#define WMMA_F16(A, Bv, Cv) \
  __builtin_amdgcn_wmma_f32_16x16x32_f16(false, (A), false, (Bv), (short)0, (Cv), false, false)

// ---- wave32 cross-lane helpers (ds_bpermute; index is byte address lane*4) ----
__device__ __forceinline__ float xor_f(float v, int d) {
  const int l = (int)(threadIdx.x & 31u);
  return __int_as_float(__builtin_amdgcn_ds_bpermute(((l ^ d) << 2), __float_as_int(v)));
}
__device__ __forceinline__ int xor_i(int v, int d) {
  const int l = (int)(threadIdx.x & 31u);
  return __builtin_amdgcn_ds_bpermute(((l ^ d) << 2), v);
}

// ---------------------------------------------------------------------------
// Kernel 1: L2-normalize each query row, convert to f16.
// ---------------------------------------------------------------------------
__global__ __launch_bounds__(128) void k_norm_x(const float* __restrict__ x,
                                                _Float16* __restrict__ xh) {
  const int row = blockIdx.x;
  const int tid = threadIdx.x;
  __shared__ float red[4];
  const float* xr = x + (size_t)row * D_K;
  float s = 0.f;
  for (int i = tid; i < D_K; i += 128) { const float v = xr[i]; s = fmaf(v, v, s); }
#pragma unroll
  for (int d = 16; d > 0; d >>= 1) s += xor_f(s, d);
  if ((tid & 31) == 0) red[tid >> 5] = s;
  __syncthreads();
  const float tot = red[0] + red[1] + red[2] + red[3];
  const float rn  = 1.0f / fmaxf(sqrtf(tot), 1e-12f);
  _Float16* xo = xh + (size_t)row * D_K;
  for (int i = tid; i < D_K; i += 128) xo[i] = (_Float16)(xr[i] * rn);
}

// ---------------------------------------------------------------------------
// Kernel 2: memory bank fp32 -> f16 (vectorized); 64 MB result lives in L2.
// ---------------------------------------------------------------------------
__global__ __launch_bounds__(256) void k_cvt_mem(const float* __restrict__ m,
                                                 _Float16* __restrict__ mh) {
  const size_t i = ((size_t)blockIdx.x * 256 + threadIdx.x) * 4;
  const float4 v = *(const float4*)(m + i);
  v4h h = { (_Float16)v.x, (_Float16)v.y, (_Float16)v.z, (_Float16)v.w };
  *(v4h*)(mh + i) = h;
}

// ---------------------------------------------------------------------------
// Kernel 3: WMMA GEMM (dist tile) fused with streaming per-row top-32.
// Grid: x = 32 column chunks (2048 cols), y = 16 row blocks (64 rows).
// 8 waves = 2 M-waves (32 rows ea) x 4 N-waves (64 cols ea); each wave holds
// 2x4 = 8 v8f accumulators (32x64 outputs), so every B fragment feeds 2 WMMAs.
// ---------------------------------------------------------------------------
__global__ __launch_bounds__(256) void k_gemm_topk(const _Float16* __restrict__ xh,
                                                   const _Float16* __restrict__ mh,
                                                   float* __restrict__ cand_val,
                                                   int* __restrict__ cand_idx) {
  __shared__ float tile[BM][TILE_N + 4];
  __shared__ float topv[BM][K_TOP];
  __shared__ int   topi[BM][K_TOP];

  const int tid  = threadIdx.x;
  const int lane = tid & 31;
  const int wave = tid >> 5;
  const int mw   = wave & 1;   // 0..1 -> 32-row strip
  const int nw   = wave >> 1;  // 0..3 -> 64-col strip
  const int row0 = blockIdx.y * BM;
  const int chunk = blockIdx.x;

  for (int i = tid; i < BM * K_TOP; i += 256) {
    topv[i >> 5][i & 31] = NEG_INF;
    topi[i >> 5][i & 31] = 0;
  }

  // A fragment addressing (16x32 f16): lane = M (0..15) | K-half (lane>>4)
  const int am = lane & 15;
  const int ak = lane >> 4;
  const _Float16* aRow0 = xh + (size_t)(row0 + mw * 32 + am) * D_K;       // M-tile 0
  const _Float16* aRow1 = aRow0 + (size_t)16 * D_K;                       // M-tile 1
  // B fragment addressing (32x16 f16): lane = N (0..15) | K-group (lane>>4)
  const int bn = lane & 15;
  const int bk = lane >> 4;

  for (int t = 0; t < TILES_PER_CHUNK; ++t) {
    const int n0 = chunk * CHUNK_COLS + t * TILE_N;
    v8f acc00 = {}, acc01 = {}, acc02 = {}, acc03 = {};
    v8f acc10 = {}, acc11 = {}, acc12 = {}, acc13 = {};
    const _Float16* b0 = mh + (size_t)(n0 + nw * 64 +  0 + bn) * D_K + bk * 16;
    const _Float16* b1 = mh + (size_t)(n0 + nw * 64 + 16 + bn) * D_K + bk * 16;
    const _Float16* b2 = mh + (size_t)(n0 + nw * 64 + 32 + bn) * D_K + bk * 16;
    const _Float16* b3 = mh + (size_t)(n0 + nw * 64 + 48 + bn) * D_K + bk * 16;
#pragma unroll 4
    for (int kk = 0; kk < D_K; kk += 32) {
      const v8h a0lo = *(const v8h*)(aRow0 + kk + ak * 16);
      const v8h a0hi = *(const v8h*)(aRow0 + kk + ak * 16 + 8);
      const v16h a0 = __builtin_shufflevector(a0lo, a0hi,
          0, 1, 2, 3, 4, 5, 6, 7, 8, 9, 10, 11, 12, 13, 14, 15);
      const v8h a1lo = *(const v8h*)(aRow1 + kk + ak * 16);
      const v8h a1hi = *(const v8h*)(aRow1 + kk + ak * 16 + 8);
      const v16h a1 = __builtin_shufflevector(a1lo, a1hi,
          0, 1, 2, 3, 4, 5, 6, 7, 8, 9, 10, 11, 12, 13, 14, 15);

      const v16h vb0 = *(const v16h*)(b0 + kk);
      acc00 = WMMA_F16(a0, vb0, acc00);
      acc10 = WMMA_F16(a1, vb0, acc10);
      const v16h vb1 = *(const v16h*)(b1 + kk);
      acc01 = WMMA_F16(a0, vb1, acc01);
      acc11 = WMMA_F16(a1, vb1, acc11);
      const v16h vb2 = *(const v16h*)(b2 + kk);
      acc02 = WMMA_F16(a0, vb2, acc02);
      acc12 = WMMA_F16(a1, vb2, acc12);
      const v16h vb3 = *(const v16h*)(b3 + kk);
      acc03 = WMMA_F16(a0, vb3, acc03);
      acc13 = WMMA_F16(a1, vb3, acc13);
    }

    __syncthreads();  // previous tile's selection readers are done
    {
      // C layout: VGPR e -> M = e + 8*(lane>=16), N = lane&15
      const int mo = mw * 32 + (lane >> 4) * 8;
      const int no = nw * 64 + (lane & 15);
#pragma unroll
      for (int e = 0; e < 8; ++e) {
        tile[mo + e][no +  0]      = acc00[e];
        tile[mo + e][no + 16]      = acc01[e];
        tile[mo + e][no + 32]      = acc02[e];
        tile[mo + e][no + 48]      = acc03[e];
        tile[mo + 16 + e][no +  0] = acc10[e];
        tile[mo + 16 + e][no + 16] = acc11[e];
        tile[mo + 16 + e][no + 32] = acc12[e];
        tile[mo + 16 + e][no + 48] = acc13[e];
      }
    }
    __syncthreads();

    // Streaming top-32: each wave exclusively owns 8 rows. Threshold filter +
    // rare serialized sorted-insertion (at least 32 of a row's 2048 chunk columns pass).
    for (int rr = 0; rr < 8; ++rr) {
      const int r = wave * 8 + rr;
#pragma unroll
      for (int j = 0; j < TILE_N / 32; ++j) {
        const int   c  = j * 32 + lane;          // lane-consecutive: no bank conflicts
        const float v  = tile[r][c];
        const float th = topv[r][K_TOP - 1];
        unsigned msk = __builtin_amdgcn_ballot_w32(v > th);
        while (msk) {
          const int   src = __builtin_ctz(msk);
          const float cv  = __int_as_float(__builtin_amdgcn_readlane(__float_as_int(v), src));
          const int   ci  = n0 + __builtin_amdgcn_readlane(c, src);
          if (lane == 0 && cv > topv[r][K_TOP - 1]) {
            int p = K_TOP - 1;
            while (p > 0 && topv[r][p - 1] < cv) {
              topv[r][p] = topv[r][p - 1];
              topi[r][p] = topi[r][p - 1];
              --p;
            }
            topv[r][p] = cv;
            topi[r][p] = ci;
          }
          msk &= msk - 1;
        }
      }
    }
  }

  __syncthreads();
  for (int i = tid; i < BM * K_TOP; i += 256) {
    const int r = i >> 5, k = i & 31;
    const size_t o = (size_t)(row0 + r) * CAND_PER_ROW + (size_t)chunk * K_TOP + k;
    cand_val[o] = topv[r][k];
    cand_idx[o] = topi[r][k];
  }
}

// ---------------------------------------------------------------------------
// Kernel 4: per row merge 32 chunks x 32 candidates -> global top-32,
// softmax(T=0.07), label gather, LDS scatter over 345 classes, writeout.
// One wave32 per row.
// ---------------------------------------------------------------------------
__global__ __launch_bounds__(32) void k_final(const float* __restrict__ cand_val,
                                              const int* __restrict__ cand_idx,
                                              const int* __restrict__ labels,
                                              float* __restrict__ out) {
  const int row  = blockIdx.x;
  const int lane = threadIdx.x;
  __shared__ float pred[C_N];

  float cv[32];
  int   ci[32];
  const size_t base = (size_t)row * CAND_PER_ROW;
#pragma unroll
  for (int c = 0; c < 32; ++c) {
    cv[c] = cand_val[base + c * 32 + lane];
    ci[c] = cand_idx[base + c * 32 + lane];
  }
  for (int i = lane; i < C_N; i += 32) pred[i] = 0.f;

  float selv = 0.f; int seli = 0; float vmax = 0.f;
  for (int k = 0; k < K_TOP; ++k) {
    float lv = NEG_INF; int li = 0x7fffffff;
#pragma unroll
    for (int c = 0; c < 32; ++c)
      if (cv[c] > lv || (cv[c] == lv && ci[c] < li)) { lv = cv[c]; li = ci[c]; }
    float bv = lv; int bi = li;
#pragma unroll
    for (int d = 16; d > 0; d >>= 1) {
      const float ov = xor_f(bv, d);
      const int   oi = xor_i(bi, d);
      if (ov > bv || (ov == bv && oi < bi)) { bv = ov; bi = oi; }
    }
    if (k == 0) vmax = bv;
#pragma unroll
    for (int c = 0; c < 32; ++c)
      if (cv[c] == bv && ci[c] == bi) cv[c] = NEG_INF;  // unique (val,idx) pair
    if (lane == k) { selv = bv; seli = bi; }
  }

  const float e = __expf((selv - vmax) * INV_T);
  float s = e;
#pragma unroll
  for (int d = 16; d > 0; d >>= 1) s += xor_f(s, d);
  const float w = e / s;
  const int lab = labels[seli];
  __syncthreads();
  atomicAdd(&pred[lab], w);   // ds_add_f32
  __syncthreads();
  float* o = out + (size_t)row * C_N;
  for (int i = lane; i < C_N; i += 32) o[i] = fminf(pred[i] + 1e-5f, 1.0f);
}

// ---------------------------------------------------------------------------
extern "C" void kernel_launch(void* const* d_in, const int* in_sizes, int n_in,
                              void* d_out, int out_size, void* d_ws, size_t ws_size,
                              hipStream_t stream) {
  (void)in_sizes; (void)n_in; (void)out_size; (void)ws_size;
  const float* x      = (const float*)d_in[0];
  const float* mem    = (const float*)d_in[1];
  const int*   labels = (const int*)d_in[2];
  float*       out    = (float*)d_out;

  char* w = (char*)d_ws;
  _Float16* xh = (_Float16*)w;      w += (size_t)B_N * D_K * sizeof(_Float16);   // 1 MB
  _Float16* mh = (_Float16*)w;      w += (size_t)Q_N * D_K * sizeof(_Float16);   // 64 MB
  float* cand_val = (float*)w;      w += (size_t)B_N * CAND_PER_ROW * sizeof(float); // 4 MB
  int*   cand_idx = (int*)w;                                                          // 4 MB

  k_norm_x<<<B_N, 128, 0, stream>>>(x, xh);
  k_cvt_mem<<<(int)(((size_t)Q_N * D_K) / (4 * 256)), 256, 0, stream>>>(mem, mh);
  dim3 g3(N_CHUNKS, B_N / BM);
  k_gemm_topk<<<g3, 256, 0, stream>>>(xh, mh, cand_val, cand_idx);
  k_final<<<B_N, 32, 0, stream>>>(cand_val, cand_idx, labels, out);
}
